// Mine_54125177864859
// MI455X (gfx1250) — matrix operand order repaired and checked by reading, not compiled, not run on hardware
//
#include <hip/hip_runtime.h>
#include <hip/hip_bf16.h>

typedef __attribute__((ext_vector_type(16))) __bf16 v16bf;
typedef __attribute__((ext_vector_type(8)))  __bf16 v8bf;
typedef __attribute__((ext_vector_type(8)))  float  v8f;

#define B_TOT   16384
#define L_LAB   512
#define HID     768
#define TRANS   128
#define NTILES  (B_TOT / 16)   // 1024

static __device__ __forceinline__ __bf16 f2bf(float f) {
    unsigned u = __builtin_bit_cast(unsigned, f);
    unsigned r = u + 0x7FFFu + ((u >> 16) & 1u);   // round-to-nearest-even
    unsigned short h = (unsigned short)(r >> 16);
    return __builtin_bit_cast(__bf16, h);
}

static __device__ __forceinline__ v16bf cat8(v8bf lo, v8bf hi) {
    v16bf r;
#pragma unroll
    for (int i = 0; i < 8; ++i) { r[i] = lo[i]; r[i + 8] = hi[i]; }
    return r;
}

// A-matrix 16x32 bf16 tile per ISA layout: lane m = lane&15; lanes 0-15 cover
// K = kbase+[0..7] then kbase+[16..23]; lanes 16-31 cover kbase+[8..15], +[24..31].
static __device__ __forceinline__ v16bf loadA(const __bf16* rowp, int kbase, int kh) {
    const __bf16* p = rowp + kbase + kh;
    v8bf lo = *(const v8bf*)p;          // K = kbase+kh .. +7
    v8bf hi = *(const v8bf*)(p + 16);   // K = kbase+16+kh .. +7
    return cat8(lo, hi);
}

// B-matrix 32x16 tile from packed layout: pack[(kc*N + n)*32 + kk] = B[kc*32+kk][n]
// lane n = n0 + lane&15; lanes 0-15 take K=kbase+[0..7],[16..23]; lanes 16-31 K=kbase+[8..15],[24..31].
static __device__ __forceinline__ v16bf loadBp(const __bf16* pack, int N, int kc,
                                               int n0, int lane) {
    int n    = n0 + (lane & 15);
    int koff = (lane >> 4) << 3;
    const __bf16* p = pack + (((size_t)(kc * N + n)) << 5) + koff;
    v8bf lo = *(const v8bf*)p;
    v8bf hi = *(const v8bf*)(p + 16);
    return cat8(lo, hi);
}

static __device__ __forceinline__ v8f bfwmma(v16bf a, v16bf b, v8f c) {
    return __builtin_amdgcn_wmma_f32_16x16x32_bf16(false, a, false, b,
                                                   (short)0, c, false, false);
}

static __device__ __forceinline__ float softplus_f(float x) {
    float ax = fabsf(x);
    return fmaxf(x, 0.0f) + log1pf(__expf(-ax));
}

// ---------------------------------------------------------------- prep kernels
__global__ void k_zero(float* acc) { acc[0] = 0.0f; }

// src = W[n*K + k] (weight stored [N_out][K_in]); dst packed as B = W^T
__global__ void k_pack_wT(const float* __restrict__ W, __bf16* __restrict__ dst,
                          int K, int N) {
    int idx = blockIdx.x * 256 + threadIdx.x;
    if (idx >= K * N) return;
    int kk = idx & 31;
    int rest = idx >> 5;
    int n = rest % N;
    int kc = rest / N;
    dst[idx] = f2bf(W[(size_t)n * K + (kc * 32 + kk)]);
}

// src = M[k*N + n] (already B-oriented, e.g. label_embed [L][HID])
__global__ void k_pack_b(const float* __restrict__ M, __bf16* __restrict__ dst,
                         int K, int N) {
    int idx = blockIdx.x * 256 + threadIdx.x;
    if (idx >= K * N) return;
    int kk = idx & 31;
    int rest = idx >> 5;
    int n = rest % N;
    int kc = rest / N;
    dst[idx] = f2bf(M[(size_t)(kc * 32 + kk) * N + n]);
}

__global__ void k_cvt_bf(const float* __restrict__ src, __bf16* __restrict__ dst,
                         int n) {
    int i = blockIdx.x * 256 + threadIdx.x;
    if (i < n) dst[i] = f2bf(src[i]);
}

// ------------------------------------------------- pooling: (mask@labels)/cnt
__global__ void __launch_bounds__(256)
k_pool(const int* __restrict__ target, const __bf16* __restrict__ labP,
       __bf16* __restrict__ labf) {
    __shared__ __bf16 maskA[16 * L_LAB];   // 16 KB
    __shared__ float  inv[16];

    const int tid  = threadIdx.x;
    const int b0   = blockIdx.x * 16;
    const int lane = tid & 31;
    const int wave = tid >> 5;

    // stage mask tile as bf16 + per-row counts; each thread owns a contiguous
    // 32-element run of one row (16 threads per row, 16 rows).
    {
        const int trow = tid >> 4;
        const int tcol = (tid & 15) * 32;
        const int* tr = target + (size_t)(b0 + trow) * L_LAB + tcol;
        __bf16* mrow = &maskA[trow * L_LAB + tcol];
        int c = 0;
#pragma unroll 8
        for (int j = 0; j < 32; ++j) {
            int v = tr[j];
            c += v;
            mrow[j] = f2bf((float)v);
        }
#pragma unroll
        for (int off = 8; off > 0; off >>= 1) c += __shfl_xor(c, off, 16);
        if ((tid & 15) == 0) inv[trow] = 1.0f / (float)max(c, 1);
    }
    __syncthreads();

    const int m  = lane & 15;
    const int kh = (lane >> 4) << 3;
    const int mb = (lane >> 4) << 3;
    const __bf16* rowp = &maskA[m * L_LAB];

    // each wave owns 3 pairs of adjacent 16-wide N-tiles: 2 independent
    // accumulator chains per pair -> loads overlap the partner chain's WMMA.
#pragma unroll 1
    for (int np = 0; np < 3; ++np) {
        const int n0 = (wave * 3 + np) * 32;
        v8f c0 = {}, c1 = {};
#pragma unroll
        for (int kc = 0; kc < L_LAB / 32; ++kc) {
            v16bf a  = loadA(rowp, kc * 32, kh);
            v16bf b0 = loadBp(labP, HID, kc, n0, lane);
            v16bf b1 = loadBp(labP, HID, kc, n0 + 16, lane);
            c0 = bfwmma(a, b0, c0);
            c1 = bfwmma(a, b1, c1);
        }
        int n = n0 + (lane & 15);
#pragma unroll
        for (int r = 0; r < 8; ++r) {
            float iv = inv[mb + r];
            size_t row = (size_t)(b0 + mb + r) * HID;
            labf[row + n]      = f2bf(c0[r] * iv);
            labf[row + n + 16] = f2bf(c1[r] * iv);
        }
    }
}

// ------------------------------------------------------------- fused MLP + loss
__global__ void __launch_bounds__(256)
k_mlp(const __bf16* __restrict__ textB, const __bf16* __restrict__ labfB,
      const int* __restrict__ perm,
      const __bf16* __restrict__ WtP, const __bf16* __restrict__ WlP,
      const __bf16* __restrict__ W0P, const __bf16* __restrict__ W1P,
      const float* __restrict__ b_text, const float* __restrict__ b_label,
      const float* __restrict__ b0v, const float* __restrict__ b1v,
      const float* __restrict__ W2, const float* __restrict__ b2v,
      float* __restrict__ acc) {
    __shared__ __bf16 hpos[16 * 256], hneg[16 * 256];   // 8 KB each
    __shared__ __bf16 h2p[16 * 128], h2n[16 * 128];     // 4 KB each
    __shared__ __bf16 h3p[16 * 128], h3n[16 * 128];     // 4 KB each

    const int tid  = threadIdx.x;
    const int lane = tid & 31;
    const int wave = tid >> 5;           // 8 waves -> 8 N-tiles of 16
    const int b0   = blockIdx.x * 16;
    const int m    = lane & 15;
    const int kh   = (lane >> 4) << 3;
    const int mb   = (lane >> 4) << 3;
    const int nt   = wave;
    const int ng   = nt * 16 + (lane & 15);

    // ---- stage 1: tt(pos), tt(neg), ll -> h[16][256] in LDS (3 indep chains)
    const __bf16* tp = textB + (size_t)(b0 + m) * HID;
    const __bf16* tn = textB + (size_t)perm[b0 + m] * HID;
    const __bf16* lf = labfB + (size_t)(b0 + m) * HID;

    v8f ct = {}, cn = {}, cl = {};
#pragma unroll
    for (int kc = 0; kc < HID / 32; ++kc) {
        int kb = kc * 32;
        v16bf bw = loadBp(WtP, TRANS, kc, nt * 16, lane);
        v16bf bl = loadBp(WlP, TRANS, kc, nt * 16, lane);
        v16bf at = loadA(tp, kb, kh);
        v16bf an = loadA(tn, kb, kh);
        v16bf al = loadA(lf, kb, kh);
        ct = bfwmma(at, bw, ct);
        cn = bfwmma(an, bw, cn);
        cl = bfwmma(al, bl, cl);
    }
    {
        float bt = b_text[ng], blb = b_label[ng];
#pragma unroll
        for (int r = 0; r < 8; ++r) {
            int row = mb + r;
            hpos[row * 256 + ng] = f2bf(ct[r] + bt);
            hneg[row * 256 + ng] = f2bf(cn[r] + bt);
            __bf16 lv = f2bf(cl[r] + blb);
            hpos[row * 256 + 128 + ng] = lv;
            hneg[row * 256 + 128 + ng] = lv;
        }
    }
    __syncthreads();

    // ---- stage 2: relu(h @ W0^T + b0) -> h2[16][128] (2 indep chains)
    {
        v8f cp = {}, cg = {};
        const __bf16* hp = &hpos[m * 256];
        const __bf16* hn = &hneg[m * 256];
#pragma unroll
        for (int kc = 0; kc < 256 / 32; ++kc) {
            v16bf bw = loadBp(W0P, TRANS, kc, nt * 16, lane);
            v16bf ap = loadA(hp, kc * 32, kh);
            v16bf ag = loadA(hn, kc * 32, kh);
            cp = bfwmma(ap, bw, cp);
            cg = bfwmma(ag, bw, cg);
        }
        float bb = b0v[ng];
#pragma unroll
        for (int r = 0; r < 8; ++r) {
            int row = mb + r;
            h2p[row * 128 + ng] = f2bf(fmaxf(cp[r] + bb, 0.0f));
            h2n[row * 128 + ng] = f2bf(fmaxf(cg[r] + bb, 0.0f));
        }
    }
    __syncthreads();

    // ---- stage 3: relu(h2 @ W1^T + b1) -> h3[16][128] (2 indep chains)
    {
        v8f cp = {}, cg = {};
        const __bf16* hp = &h2p[m * 128];
        const __bf16* hn = &h2n[m * 128];
#pragma unroll
        for (int kc = 0; kc < 128 / 32; ++kc) {
            v16bf bw = loadBp(W1P, TRANS, kc, nt * 16, lane);
            v16bf ap = loadA(hp, kc * 32, kh);
            v16bf ag = loadA(hn, kc * 32, kh);
            cp = bfwmma(ap, bw, cp);
            cg = bfwmma(ag, bw, cg);
        }
        float bb = b1v[ng];
#pragma unroll
        for (int r = 0; r < 8; ++r) {
            int row = mb + r;
            h3p[row * 128 + ng] = f2bf(fmaxf(cp[r] + bb, 0.0f));
            h3n[row * 128 + ng] = f2bf(fmaxf(cg[r] + bb, 0.0f));
        }
    }
    __syncthreads();

    // ---- stage 4: net = h3 @ W2 + b2; loss terms; block reduce; atomicAdd
    if (wave == 0) {
        bool pos = lane < 16;
        const __bf16* hr = (pos ? h3p : h3n) + (lane & 15) * 128;
        float s = b2v[0];
#pragma unroll 8
        for (int k = 0; k < 128; ++k) s += (float)hr[k] * W2[k];
        float sp = pos ? softplus_f(-s) : softplus_f(s);
#pragma unroll
        for (int off = 16; off > 0; off >>= 1) sp += __shfl_xor(sp, off, 32);
        if (lane == 0) atomicAdd(acc, sp);
    }
}

__global__ void k_final(const float* __restrict__ acc, float* __restrict__ out) {
    out[0] = acc[0] * (1.0f / (float)B_TOT);
}

// --------------------------------------------------------------------- launch
extern "C" void kernel_launch(void* const* d_in, const int* in_sizes, int n_in,
                              void* d_out, int out_size, void* d_ws, size_t ws_size,
                              hipStream_t stream) {
    const float* text  = (const float*)d_in[0];
    const float* label = (const float*)d_in[1];
    const int*   targ  = (const int*)d_in[2];
    const int*   perm  = (const int*)d_in[3];
    const float* W_text = (const float*)d_in[4];
    const float* b_text = (const float*)d_in[5];
    const float* W_label = (const float*)d_in[6];
    const float* b_label = (const float*)d_in[7];
    const float* W0 = (const float*)d_in[8];
    const float* b0 = (const float*)d_in[9];
    const float* W1 = (const float*)d_in[10];
    const float* b1 = (const float*)d_in[11];
    const float* W2 = (const float*)d_in[12];
    const float* b2 = (const float*)d_in[13];
    float* out = (float*)d_out;

    char* ws = (char*)d_ws;
    size_t o = 0;
    float*  acc  = (float*)(ws + o);  o += 256;
    __bf16* WtP  = (__bf16*)(ws + o); o += (size_t)HID * TRANS * 2;       // 192 KB
    __bf16* WlP  = (__bf16*)(ws + o); o += (size_t)HID * TRANS * 2;       // 192 KB
    __bf16* W0P  = (__bf16*)(ws + o); o += (size_t)256 * TRANS * 2;       //  64 KB
    __bf16* W1P  = (__bf16*)(ws + o); o += (size_t)128 * TRANS * 2;       //  32 KB
    __bf16* LabP = (__bf16*)(ws + o); o += (size_t)L_LAB * HID * 2;       // 768 KB
    __bf16* txB  = (__bf16*)(ws + o); o += (size_t)B_TOT * HID * 2;       //  24 MB
    __bf16* lfB  = (__bf16*)(ws + o); o += (size_t)B_TOT * HID * 2;       //  24 MB

    k_zero<<<1, 1, 0, stream>>>(acc);
    k_pack_wT<<<(HID * TRANS) / 256, 256, 0, stream>>>(W_text, WtP, HID, TRANS);
    k_pack_wT<<<(HID * TRANS) / 256, 256, 0, stream>>>(W_label, WlP, HID, TRANS);
    k_pack_wT<<<(256 * TRANS) / 256, 256, 0, stream>>>(W0, W0P, 256, TRANS);
    k_pack_wT<<<(128 * TRANS) / 256, 256, 0, stream>>>(W1, W1P, 128, TRANS);
    k_pack_b<<<(L_LAB * HID) / 256, 256, 0, stream>>>(label, LabP, L_LAB, HID);
    k_cvt_bf<<<(B_TOT * HID) / 256, 256, 0, stream>>>(text, txB, B_TOT * HID);
    k_pool<<<NTILES, 256, 0, stream>>>(targ, LabP, lfB);
    k_mlp<<<NTILES, 256, 0, stream>>>(txB, lfB, perm, WtP, WlP, W0P, W1P,
                                      b_text, b_label, b0, b1, W2, b2, acc);
    k_final<<<1, 1, 0, stream>>>(acc, out);
}
